// CausalSelfAttention_23398981829316
// MI455X (gfx1250) — matrix operand order repaired and checked by reading, not compiled, not run on hardware
//
#include <hip/hip_runtime.h>

#define B_ 2
#define S_ 2048
#define E_ 2048
#define H_ 16
#define D_ 128
#define M_ (B_ * S_)      // 4096 rows of x
#define NQKV_ (3 * E_)    // 6144

typedef __bf16 bf16_t;
typedef __attribute__((ext_vector_type(16))) bf16_t v16bf;
typedef __attribute__((ext_vector_type(8)))  bf16_t v8bf;
typedef __attribute__((ext_vector_type(8)))  float  v8f;
typedef __attribute__((ext_vector_type(4)))  unsigned int v4u;
typedef __attribute__((ext_vector_type(8)))  int v8i;
typedef __attribute__((ext_vector_type(4)))  int v4i;

#if defined(__has_builtin)
#if __has_builtin(__builtin_amdgcn_tensor_load_to_lds) && \
    __has_builtin(__builtin_amdgcn_s_wait_tensorcnt)
#define USE_TDM 1
#endif
#endif

__device__ __forceinline__ unsigned short f2bf(float f) {
  union { float f; unsigned u; } cv; cv.f = f;
  unsigned u = cv.u;
  u += 0x7FFFu + ((u >> 16) & 1u);          // round-to-nearest-even
  return (unsigned short)(u >> 16);
}
__device__ __forceinline__ v8bf  ld8 (const unsigned short* p) { return *(const v8bf *)p; }
__device__ __forceinline__ v16bf ld16(const unsigned short* p) { return *(const v16bf*)p; }
__device__ __forceinline__ v16bf cat16(v8bf lo, v8bf hi) {
  return __builtin_shufflevector(lo, hi, 0,1,2,3,4,5,6,7,8,9,10,11,12,13,14,15);
}

#ifdef USE_TDM
// 2D bf16 tile DMA: global (row-major, row_stride elems) -> LDS (contiguous).
// D# layout per CDNA5 ISA 8.3-8.5: group0 {count,lds_addr,global_addr,type=2},
// group1 {data_size=1(2B), tensor_dim0/1, tile_dim0/1, tensor_dim0_stride}.
// 6-arg builtin on this toolchain: (g0, g1, g2, g3, g4, cpol).
__device__ __forceinline__ void tdm_load_2d(unsigned lds_byte_addr,
                                            const unsigned short* gptr,
                                            unsigned tile_cols, unsigned tile_rows,
                                            unsigned tensor_cols, unsigned tensor_rows,
                                            unsigned row_stride) {
  unsigned long long ga = (unsigned long long)(uintptr_t)gptr;
  v4u g0;
  g0[0] = 1u;                                              // count=1 (user D#)
  g0[1] = lds_byte_addr;                                   // lds_addr
  g0[2] = (unsigned)(ga & 0xFFFFFFFFu);                    // global_addr lo
  g0[3] = (unsigned)((ga >> 32) & 0x1FFFFFFu) | (2u << 30);// addr hi | type=2
  v8i g1;
  g1[0] = (int)(1u << 16);                                 // data_size=1 (2B)
  g1[1] = (int)((tensor_cols & 0xFFFFu) << 16);            // tensor_dim0 lo16
  g1[2] = (int)((tensor_cols >> 16) | ((tensor_rows & 0xFFFFu) << 16));
  g1[3] = (int)((tensor_rows >> 16) | (tile_cols << 16));  // tile_dim0
  g1[4] = (int)(tile_rows & 0xFFFFu);                      // tile_dim1
  g1[5] = (int)row_stride;                                 // dim0_stride lo32
  g1[6] = 0;
  g1[7] = 0;
  v4i g2; g2[0]=0; g2[1]=0; g2[2]=0; g2[3]=0;              // 2D: groups 2/3 unused
  v4i g3; g3[0]=0; g3[1]=0; g3[2]=0; g3[3]=0;
  v8i g4; g4[0]=0; g4[1]=0; g4[2]=0; g4[3]=0; g4[4]=0; g4[5]=0; g4[6]=0; g4[7]=0;
  __builtin_amdgcn_tensor_load_to_lds(g0, g1, g2, g3, g4, 0);
}
#endif

// ---------------------------------------------------------------------------
// fp32 -> bf16 pre-pass (one shot over x / Wqkv / Wout)
// ---------------------------------------------------------------------------
__global__ __launch_bounds__(256)
void cvt_bf16(const float* __restrict__ src, unsigned short* __restrict__ dst, int n) {
  int i = (blockIdx.x * 256 + threadIdx.x) * 8;
  if (i >= n) return;
  float4 a = *(const float4*)(src + i);
  float4 b = *(const float4*)(src + i + 4);
  union { unsigned short us[8]; uint4 v; } p;
  p.us[0]=f2bf(a.x); p.us[1]=f2bf(a.y); p.us[2]=f2bf(a.z); p.us[3]=f2bf(a.w);
  p.us[4]=f2bf(b.x); p.us[5]=f2bf(b.y); p.us[6]=f2bf(b.z); p.us[7]=f2bf(b.w);
  *(uint4*)(dst + i) = p.v;
}

// ---------------------------------------------------------------------------
// RoPE cos/sin cache: [S][64] each. angle(pos, j) = pos * 10000^(-2j/128)
// (reference indexes cos/sin by d with j = d % 64)
// ---------------------------------------------------------------------------
__global__ __launch_bounds__(256)
void rope_table(float* __restrict__ cost, float* __restrict__ sint) {
  int i   = blockIdx.x * 256 + threadIdx.x;   // over S_*64
  int pos = i >> 6, j = i & 63;
  float inv = __expf(-(float)(2 * j) * (1.0f / 128.0f) * 9.210340371976184f);
  float ang = (float)pos * inv;
  cost[i] = cosf(ang);
  sint[i] = sinf(ang);
}

// ---------------------------------------------------------------------------
// Tiled GEMM: C[M x N] = A[M x K] * W[N x K]^T, A/W bf16 in global.
// Tiles staged global->LDS by the Tensor Data Mover (double buffered).
// MODE 0: f32 store to C. MODE 1: QKV epilogue, fused RoPE (table), bf16 split.
// Block: 256 threads (8 waves), tile 128(M) x 64(N), K-step 32.
// ---------------------------------------------------------------------------
template <int MODE>
__global__ __launch_bounds__(256)
void gemm_wmma(const unsigned short* __restrict__ A, const unsigned short* __restrict__ W,
               float* __restrict__ C, int N, int K,
               unsigned short* __restrict__ qp, unsigned short* __restrict__ kp,
               unsigned short* __restrict__ vp,
               const float* __restrict__ cost, const float* __restrict__ sint) {
  __shared__ unsigned short As[2][128 * 32];   // 2 x 8 KB
  __shared__ unsigned short Bs[2][64 * 32];    // 2 x 4 KB

  const int tid    = threadIdx.x;
  const int lane   = tid & 31;
  const int w      = tid >> 5;
  const int lane15 = lane & 15;
  const bool hih   = lane >= 16;
  const int m0     = blockIdx.y * 128;
  const int n0     = blockIdx.x * 64;

  v8f acc[4];
  #pragma unroll
  for (int i = 0; i < 4; ++i)
    #pragma unroll
    for (int j = 0; j < 8; ++j) acc[i][j] = 0.f;

#ifdef USE_TDM
  // prologue: DMA tile 0 into buffer 0 (wave0: A-tile, wave1: B-tile)
  if (w == 0)
    tdm_load_2d((unsigned)(uintptr_t)&As[0][0], A + (size_t)m0 * K,
                32, 128, K, M_, K);
  if (w == 1)
    tdm_load_2d((unsigned)(uintptr_t)&Bs[0][0], W + (size_t)n0 * K,
                32, 64, K, N, K);
#endif

  for (int it = 0, k0 = 0; k0 < K; k0 += 32, ++it) {
    const int cur = it & 1;
#ifdef USE_TDM
    __builtin_amdgcn_s_wait_tensorcnt(0);
    __syncthreads();                       // tile `cur` visible to all waves
    if (k0 + 32 < K) {                     // prefetch next tile into cur^1
      if (w == 0)
        tdm_load_2d((unsigned)(uintptr_t)&As[cur ^ 1][0],
                    A + (size_t)m0 * K + k0 + 32, 32, 128, K, M_, K);
      if (w == 1)
        tdm_load_2d((unsigned)(uintptr_t)&Bs[cur ^ 1][0],
                    W + (size_t)n0 * K + k0 + 32, 32, 64, K, N, K);
    }
#else
    __syncthreads();
    { // cooperative copy, bf16 -> bf16 (uint4 = 8 elems)
      int r = tid >> 1, seg = (tid & 1) * 16;
      const uint4* src = (const uint4*)(A + (size_t)(m0 + r) * K + k0 + seg);
      uint4* dst = (uint4*)&As[cur][r * 32 + seg];
      dst[0] = src[0]; dst[1] = src[1];
    }
    if (tid < 128) {
      int r = tid >> 1, seg = (tid & 1) * 16;
      const uint4* src = (const uint4*)(W + (size_t)(n0 + r) * K + k0 + seg);
      uint4* dst = (uint4*)&Bs[cur][r * 32 + seg];
      dst[0] = src[0]; dst[1] = src[1];
    }
    __syncthreads();
#endif
    { // wave w owns rows [w*16, w*16+16)
      int m  = w * 16 + lane15;
      int kb = hih ? 16 : 0;
      v16bf a = cat16(ld8(&As[cur][m * 32 + kb]), ld8(&As[cur][m * 32 + kb + 8]));
      #pragma unroll
      for (int nt = 0; nt < 4; ++nt) {
        v16bf b = ld16(&Bs[cur][(nt * 16 + lane15) * 32 + (hih ? 16 : 0)]);
        acc[nt] = __builtin_amdgcn_wmma_f32_16x16x32_bf16(
            false, a, false, b, (short)0, acc[nt], false, false);
      }
    }
  }

  if (MODE == 0) {
    #pragma unroll
    for (int nt = 0; nt < 4; ++nt) {
      int col = n0 + nt * 16 + lane15;
      #pragma unroll
      for (int r = 0; r < 8; ++r) {
        int row = m0 + w * 16 + (hih ? r + 8 : r);
        C[(size_t)row * N + col] = acc[nt][r];
      }
    }
  } else {
    #pragma unroll
    for (int nt = 0; nt < 4; ++nt) {
      int col  = n0 + nt * 16 + lane15;
      int g    = col >> 11;          // 0=q 1=k 2=v
      int rem  = col & 2047;
      int h    = rem >> 7;
      int d    = rem & 127;
      int didx = d & 63;
      #pragma unroll
      for (int r = 0; r < 8; ++r) {
        int row = m0 + w * 16 + (hih ? r + 8 : r);
        int b   = row >> 11;
        int s   = row & 2047;
        float val  = acc[nt][r];
        float pair = __shfl_xor(val, 1, 32);   // neighboring column d^1
        float res  = val;
        if (g < 2) {
          float c  = cost[s * 64 + didx];
          float sn = sint[s * 64 + didx];
          res = (d & 1) ? (val * c + pair * sn) : (val * c - pair * sn);
        }
        unsigned short* dstp = (g == 0) ? qp : (g == 1) ? kp : vp;
        dstp[(((size_t)(b * H_ + h)) * S_ + s) * D_ + d] = f2bf(res);
      }
    }
  }
}

// ---------------------------------------------------------------------------
// Flash attention, one workgroup per (128 query rows, head).
// q/k/v bf16 (B,H,S,D); output bf16 (B,S,H*D) for the out-projection.
// ---------------------------------------------------------------------------
__global__ __launch_bounds__(256)
void flash_attn(const unsigned short* __restrict__ qg, const unsigned short* __restrict__ kg,
                const unsigned short* __restrict__ vg, unsigned short* __restrict__ og) {
  __shared__ unsigned short Qs[128 * 128];   // 32 KB
  __shared__ unsigned short Ks[64 * 128];    // 16 KB
  __shared__ unsigned short Vt[128 * 64];    // 16 KB (transposed: [d][k])
  __shared__ unsigned short Ps[8 * 16 * 64]; // 16 KB (per-wave P tiles)

  const int tid    = threadIdx.x;
  const int lane   = tid & 31;
  const int w      = tid >> 5;
  const int lane15 = lane & 15;
  const bool hih   = lane >= 16;
  const int q0     = blockIdx.x * 128;
  const int bh     = blockIdx.y;
  const int b      = bh >> 4, h = bh & 15;
  const size_t base = (size_t)bh * S_ * D_;
  const float NEGINF = -__builtin_inff();

  { // stage Q tile (pure bf16 copy)
    int r = tid >> 1, d0 = (tid & 1) * 64;
    const uint4* src = (const uint4*)(qg + base + (size_t)(q0 + r) * D_ + d0);
    uint4* dst = (uint4*)&Qs[r * 128 + d0];
    #pragma unroll
    for (int i = 0; i < 8; ++i) dst[i] = src[i];
  }

  v8f oacc[8];
  #pragma unroll
  for (int j = 0; j < 8; ++j)
    #pragma unroll
    for (int r = 0; r < 8; ++r) oacc[j][r] = 0.f;
  float m_s[8], l_s[8];
  #pragma unroll
  for (int r = 0; r < 8; ++r) { m_s[r] = NEGINF; l_s[r] = 0.f; }

  const int ktiles = (q0 + 128) >> 6;        // causal trim
  for (int kt = 0; kt < ktiles; ++kt) {
    const int kb = kt * 64;
    __syncthreads();
    { // K tile 64x128 (copy)
      int r = tid >> 2, d0 = (tid & 3) * 32;
      const uint4* src = (const uint4*)(kg + base + (size_t)(kb + r) * D_ + d0);
      uint4* dst = (uint4*)&Ks[r * 128 + d0];
      #pragma unroll
      for (int i = 0; i < 4; ++i) dst[i] = src[i];
    }
    { // V tile 64x128 -> transposed Vt[d][k]
      int kk = tid >> 2, d0 = (tid & 3) * 32;
      const uint4* src = (const uint4*)(vg + base + (size_t)(kb + kk) * D_ + d0);
      #pragma unroll
      for (int i = 0; i < 4; ++i) {
        uint4 v = src[i];
        int dd = d0 + i * 8;
        Vt[(dd+0)*64+kk]=(unsigned short)(v.x & 0xFFFF); Vt[(dd+1)*64+kk]=(unsigned short)(v.x >> 16);
        Vt[(dd+2)*64+kk]=(unsigned short)(v.y & 0xFFFF); Vt[(dd+3)*64+kk]=(unsigned short)(v.y >> 16);
        Vt[(dd+4)*64+kk]=(unsigned short)(v.z & 0xFFFF); Vt[(dd+5)*64+kk]=(unsigned short)(v.z >> 16);
        Vt[(dd+6)*64+kk]=(unsigned short)(v.w & 0xFFFF); Vt[(dd+7)*64+kk]=(unsigned short)(v.w >> 16);
      }
    }
    __syncthreads();

    // S = Q * K^T (16 rows per wave x 64 keys), K-dim 128
    v8f sacc[4];
    #pragma unroll
    for (int nt = 0; nt < 4; ++nt)
      #pragma unroll
      for (int r = 0; r < 8; ++r) sacc[nt][r] = 0.f;
    #pragma unroll
    for (int e0 = 0; e0 < 128; e0 += 32) {
      int m = w * 16 + lane15;
      int cb = e0 + (hih ? 16 : 0);
      v16bf a = cat16(ld8(&Qs[m * 128 + cb]), ld8(&Qs[m * 128 + cb + 8]));
      #pragma unroll
      for (int nt = 0; nt < 4; ++nt) {
        v16bf bf = ld16(&Ks[(nt * 16 + lane15) * 128 + e0 + (hih ? 16 : 0)]);
        sacc[nt] = __builtin_amdgcn_wmma_f32_16x16x32_bf16(
            false, a, false, bf, (short)0, sacc[nt], false, false);
      }
    }

    // online softmax (row stats replicated across each 16-lane half)
    const float scale = 0.08838834764831845f;   // 1/sqrt(128)
    #pragma unroll
    for (int r = 0; r < 8; ++r) {
      int qrow = q0 + w * 16 + (hih ? r + 8 : r);
      float sv[4]; float tm = NEGINF;
      #pragma unroll
      for (int nt = 0; nt < 4; ++nt) {
        int kcol = kb + nt * 16 + lane15;
        float s = sacc[nt][r] * scale;
        if (kcol > qrow) s = NEGINF;            // causal mask
        sv[nt] = s; tm = fmaxf(tm, s);
      }
      #pragma unroll
      for (int off = 1; off < 16; off <<= 1) tm = fmaxf(tm, __shfl_xor(tm, off, 32));
      float mn    = fmaxf(m_s[r], tm);
      float alpha = __expf(m_s[r] - mn);
      float rs    = 0.f;
      int prow    = hih ? r + 8 : r;
      #pragma unroll
      for (int nt = 0; nt < 4; ++nt) {
        float p = __expf(sv[nt] - mn);
        rs += p;
        Ps[w * 1024 + prow * 64 + nt * 16 + lane15] = f2bf(p);
      }
      #pragma unroll
      for (int off = 1; off < 16; off <<= 1) rs += __shfl_xor(rs, off, 32);
      l_s[r] = l_s[r] * alpha + rs;
      m_s[r] = mn;
      #pragma unroll
      for (int j = 0; j < 8; ++j) oacc[j][r] *= alpha;
    }

    // O += P * V   (P: 16x64 per wave, V: 64x128)
    #pragma unroll
    for (int e0 = 0; e0 < 64; e0 += 32) {
      int cb = e0 + (hih ? 16 : 0);
      v16bf a = cat16(ld8(&Ps[w * 1024 + lane15 * 64 + cb]),
                      ld8(&Ps[w * 1024 + lane15 * 64 + cb + 8]));
      #pragma unroll
      for (int j = 0; j < 8; ++j) {
        v16bf bf = ld16(&Vt[(j * 16 + lane15) * 64 + e0 + (hih ? 16 : 0)]);
        oacc[j] = __builtin_amdgcn_wmma_f32_16x16x32_bf16(
            false, a, false, bf, (short)0, oacc[j], false, false);
      }
    }
  }

  // normalize and write (B,S,H*D) bf16
  float linv[8];
  #pragma unroll
  for (int r = 0; r < 8; ++r) linv[r] = 1.0f / l_s[r];
  #pragma unroll
  for (int j = 0; j < 8; ++j) {
    int d = j * 16 + lane15;
    #pragma unroll
    for (int r = 0; r < 8; ++r) {
      int srow = q0 + w * 16 + (hih ? r + 8 : r);
      og[((size_t)(b * S_ + srow)) * E_ + h * D_ + d] = f2bf(oacc[j][r] * linv[r]);
    }
  }
}

// ---------------------------------------------------------------------------
extern "C" void kernel_launch(void* const* d_in, const int* in_sizes, int n_in,
                              void* d_out, int out_size, void* d_ws, size_t ws_size,
                              hipStream_t stream) {
  (void)in_sizes; (void)n_in; (void)out_size; (void)ws_size;
  const float* x    = (const float*)d_in[0];
  const float* Wqkv = (const float*)d_in[1];
  const float* Wout = (const float*)d_in[2];
  float* out = (float*)d_out;

  const size_t TSZ = (size_t)B_ * H_ * S_ * D_;   // 8.39M elems
  unsigned short* xb    = (unsigned short*)d_ws;
  unsigned short* wqkvb = xb    + (size_t)M_ * E_;
  unsigned short* woutb = wqkvb + (size_t)NQKV_ * E_;
  unsigned short* q     = woutb + (size_t)E_ * E_;
  unsigned short* k     = q + TSZ;
  unsigned short* v     = k + TSZ;
  unsigned short* o     = v + TSZ;                 // (B,S,E) bf16
  float* cost = (float*)(o + (size_t)M_ * E_);     // [S][64]
  float* sint = cost + (size_t)S_ * 64;

  // 0) one-shot fp32 -> bf16 conversions + RoPE cache
  cvt_bf16<<<(M_ * E_)     / (8 * 256), 256, 0, stream>>>(x,    xb,    M_ * E_);
  cvt_bf16<<<(NQKV_ * E_)  / (8 * 256), 256, 0, stream>>>(Wqkv, wqkvb, NQKV_ * E_);
  cvt_bf16<<<(E_ * E_)     / (8 * 256), 256, 0, stream>>>(Wout, woutb, E_ * E_);
  rope_table<<<(S_ * 64) / 256, 256, 0, stream>>>(cost, sint);
  // 1) QKV projection + RoPE + split (bf16 out)
  gemm_wmma<1><<<dim3(NQKV_ / 64, M_ / 128), 256, 0, stream>>>(
      xb, wqkvb, nullptr, NQKV_, E_, q, k, v, cost, sint);
  // 2) causal flash attention
  flash_attn<<<dim3(S_ / 128, B_ * H_), 256, 0, stream>>>(q, k, v, o);
  // 3) output projection (f32 out)
  gemm_wmma<0><<<dim3(E_ / 64, M_ / 128), 256, 0, stream>>>(
      o, woutb, out, E_, E_, nullptr, nullptr, nullptr, nullptr, nullptr);
}
